// EmformerBlock_72980084293739
// MI455X (gfx1250) — matrix operand order repaired and checked by reading, not compiled, not run on hardware
//
#include <hip/hip_runtime.h>
#include <hip/hip_bf16.h>

// ---------------- problem constants ----------------
constexpr int B_    = 4;
constexpr int S_    = 2048;
constexpr int D_    = 512;
constexpr int H_    = 8;
constexpr int DK_   = 64;
constexpr int FFN_  = 2048;
constexpr int LEFT_ = 512;
constexpr int CHUNK_= 128;
constexpr int RIGHT_= 128;
constexpr int M_    = B_ * S_;          // 8192 rows
constexpr float EPS_ = 1e-5f;

// LDS strides (halves), padded for bank spread; *2 bytes must be 16/32B aligned
constexpr int VSTR = 784;   // Vt rows   (784*2 = 1568 B, 32B-aligned)
constexpr int SSTR = 776;   // score rows(776*2 = 1552 B, 16B-aligned)

// attention LDS layout (dynamic): Vt | scores (raw-V staging overlaps) | invsum
constexpr size_t VT_BYTES  = (size_t)64 * VSTR * 2;    // 100352
constexpr size_t SC_BYTES  = (size_t)128 * SSTR * 2;   // 198656
constexpr size_t INV_OFF   = VT_BYTES + SC_BYTES;      // 299008
constexpr size_t ATTN_LDS  = INV_OFF + 128 * 4;        // 299520 (< 320KB/WG)

typedef _Float16 half8  __attribute__((ext_vector_type(8)));
typedef _Float16 half16 __attribute__((ext_vector_type(16)));
typedef float    floatx8 __attribute__((ext_vector_type(8)));

// ---------------- WMMA helpers ----------------
__device__ __forceinline__ floatx8 wmma_f16(half16 a, half16 b, floatx8 c) {
  return __builtin_amdgcn_wmma_f32_16x16x32_f16(false, a, false, b, (short)0, c,
                                                false, false);
}

// A fragment (16x32, row-major A): lane m = lane&15 holds row m.
// halves j=0..7 -> K = kk + hi*8 ; j=8..15 -> K = kk + 16 + hi*8
__device__ __forceinline__ half16 load_a_frag(const _Float16* rowbase, int hi) {
  half8 lo = *(const half8*)(rowbase + hi * 8);
  half8 hh = *(const half8*)(rowbase + 16 + hi * 8);
  return __builtin_shufflevector(lo, hh, 0, 1, 2, 3, 4, 5, 6, 7, 8, 9, 10, 11,
                                 12, 13, 14, 15);
}

// B fragment (32x16) from an N-major ("Wt[n][k]") source: lane n = lane&15,
// halves j=0..7 -> K = kk + hi*8 ; j=8..15 -> K = kk + 16 + hi*8
__device__ __forceinline__ half16 load_b_frag(const _Float16* colbase, int hi) {
  half8 lo = *(const half8*)(colbase + hi * 8);
  half8 hh = *(const half8*)(colbase + 16 + hi * 8);
  return __builtin_shufflevector(lo, hh, 0, 1, 2, 3, 4, 5, 6, 7, 8, 9, 10, 11,
                                 12, 13, 14, 15);
}

// ---------------- weight convert + transpose (f32 [K,N] -> f16 [N,K]) -------
__global__ __launch_bounds__(256) void k_convT(const float* __restrict__ src,
                                               _Float16* __restrict__ dst,
                                               int K, int N) {
  int t = blockIdx.x * 256 + threadIdx.x;
  if (t >= K * N) return;
  int n = t % N, k = t / N;
  dst[(size_t)n * K + k] = (_Float16)src[t];
}

// ---------------- fused (A [+ Badd]) -> residual / LayerNorm -----------------
// block = 128 threads, one D_=512 row per block, 4 elems/thread
__global__ __launch_bounds__(128) void k_add_ln(
    const float* __restrict__ A, const float* __restrict__ Badd,
    const float* __restrict__ g, const float* __restrict__ be,
    float* __restrict__ res_out, float* __restrict__ out32,
    _Float16* __restrict__ out16) {
  __shared__ float s_sum[4], s_sq[4], s_stat[2];
  const size_t base = (size_t)blockIdx.x * D_;
  const int lane = threadIdx.x & 31, wave = threadIdx.x >> 5;

  float v[4];
  float s = 0.f, q = 0.f;
#pragma unroll
  for (int i = 0; i < 4; ++i) {
    int idx = threadIdx.x + i * 128;
    float x = A[base + idx];
    if (Badd) x += Badd[base + idx];
    v[i] = x;
    if (res_out) res_out[base + idx] = x;
    s += x;
    q += x * x;
  }
#pragma unroll
  for (int off = 16; off > 0; off >>= 1) {
    s += __shfl_xor(s, off, 32);
    q += __shfl_xor(q, off, 32);
  }
  if (lane == 0) { s_sum[wave] = s; s_sq[wave] = q; }
  __syncthreads();
  if (threadIdx.x == 0) {
    float ts = 0.f, tq = 0.f;
#pragma unroll
    for (int w = 0; w < 4; ++w) { ts += s_sum[w]; tq += s_sq[w]; }
    float mean = ts / (float)D_;
    float var = tq / (float)D_ - mean * mean;
    s_stat[0] = mean;
    s_stat[1] = rsqrtf(var + EPS_);
  }
  __syncthreads();
  const float mean = s_stat[0], inv = s_stat[1];
#pragma unroll
  for (int i = 0; i < 4; ++i) {
    int idx = threadIdx.x + i * 128;
    float y = (v[i] - mean) * inv * g[idx] + be[idx];
    if (out32) out32[base + idx] = y;
    if (out16) out16[base + idx] = (_Float16)y;
  }
}

// ---------------- generic f16 WMMA GEMM: C = A[M,K] @ Wt[N,K]^T + bias -------
// block = 128 threads (4 waves). Wave computes a 32(M) x 64(N) tile.
// Block tile: 32(M) x 256(N). grid = (M/32, N/256).
__global__ __launch_bounds__(128) void k_gemm_wmma(
    const _Float16* __restrict__ A, const _Float16* __restrict__ Wt,
    const float* __restrict__ bias, float* __restrict__ C32,
    _Float16* __restrict__ C16, int M, int N, int K) {
  const int lane = threadIdx.x & 31;
  const int wave = threadIdx.x >> 5;
  const int l15 = lane & 15, hi = lane >> 4;
  const int m0 = blockIdx.x * 32;
  const int n0 = blockIdx.y * 256 + wave * 64;

  floatx8 acc[2][4] = {};
  const _Float16* arow0 = A + (size_t)(m0 + l15) * K;
  const _Float16* arow1 = A + (size_t)(m0 + 16 + l15) * K;
  const _Float16* brow[4];
#pragma unroll
  for (int nt = 0; nt < 4; ++nt)
    brow[nt] = Wt + (size_t)(n0 + nt * 16 + l15) * K;

  for (int kk = 0; kk < K; kk += 32) {
    __builtin_prefetch(arow0 + kk + 128, 0, 1);  // global_prefetch_b8
    half16 a0 = load_a_frag(arow0 + kk, hi);
    half16 a1 = load_a_frag(arow1 + kk, hi);
#pragma unroll
    for (int nt = 0; nt < 4; ++nt) {
      half16 bf = load_b_frag(brow[nt] + kk, hi);
      acc[0][nt] = wmma_f16(a0, bf, acc[0][nt]);
      acc[1][nt] = wmma_f16(a1, bf, acc[1][nt]);
    }
  }
#pragma unroll
  for (int nt = 0; nt < 4; ++nt) {
    int col = n0 + nt * 16 + l15;
    float bv = bias ? bias[col] : 0.f;
#pragma unroll
    for (int mt = 0; mt < 2; ++mt) {
#pragma unroll
      for (int r = 0; r < 8; ++r) {
        int row = m0 + mt * 16 + hi * 8 + r;
        float val = acc[mt][nt][r] + bv;
        if (C32) C32[(size_t)row * N + col] = val;
        if (C16) C16[(size_t)row * N + col] = (_Float16)val;
      }
    }
  }
}

// ---------------- Emformer attention over contiguous windows -----------------
// One block (256 threads = 8 waves) handles (b, h, 128 queries = one chunk).
// Window for chunk q0: [max(0,q0-LEFT), min(S, q0+CHUNK+RIGHT)) — a multiple
// of 128 wide (<=768), dense inside, so no per-element masking is needed.
__global__ __launch_bounds__(256) void k_attn(const _Float16* __restrict__ q16,
                                              const _Float16* __restrict__ k16,
                                              const _Float16* __restrict__ v16,
                                              float* __restrict__ attnpre) {
  extern __shared__ char smem[];
  _Float16* Vt   = (_Float16*)smem;                   // [64][VSTR]
  _Float16* Sc   = (_Float16*)(smem + VT_BYTES);      // [128][SSTR]
  float*    invs = (float*)(smem + INV_OFF);          // [128]

  const int nqb = S_ / CHUNK_;                        // 16 chunks
  const int qb = blockIdx.x % nqb;
  const int hh = (blockIdx.x / nqb) % H_;
  const int bb = blockIdx.x / (nqb * H_);
  const int q0 = qb * CHUNK_;                         // == chunk start cs
  int w0 = q0 - LEFT_;              if (w0 < 0) w0 = 0;
  int w1 = q0 + CHUNK_ + RIGHT_;    if (w1 > S_) w1 = S_;
  const int L = w1 - w0;            // multiple of 128, <= 768

  // ---- phase 1a: async-stage raw V window rows into LDS (reuses Sc space) --
  // raw layout: [k][64] f16 contiguous (128 B per key row)
  {
    _Float16* raw = Sc;
    const _Float16* vbase = v16 + ((size_t)(bb * S_ + w0)) * D_ + hh * 64;
    const int nchunk = L * 8;  // number of 16-byte chunks
    for (int t = threadIdx.x; t < nchunk; t += 256) {
      int krow = t >> 3, c = t & 7;
      const _Float16* g = vbase + (size_t)krow * D_ + c * 8;
      unsigned lds_off = (unsigned)(size_t)(raw) + (unsigned)t * 16u;
      asm volatile("global_load_async_to_lds_b128 %0, %1, off"
                   :: "v"(lds_off), "v"(g)
                   : "memory");
    }
    asm volatile("s_wait_asynccnt 0" ::: "memory");
    __syncthreads();
    // ---- phase 1b: LDS->LDS transpose into Vt[d][k] ----
    for (int t = threadIdx.x; t < L * 64; t += 256) {
      int d = t & 63, k = t >> 6;
      Vt[d * VSTR + k] = raw[k * 64 + d];
    }
  }
  __syncthreads();

  const int lane = threadIdx.x & 31, wave = threadIdx.x >> 5;
  const int l15 = lane & 15, hi = lane >> 4;

  // ---- phase 2: scores = Q K^T * (1/8), into f16 LDS (unnormalized later) --
  const _Float16* qrow =
      q16 + ((size_t)(bb * S_ + q0 + wave * 16 + l15)) * D_ + hh * 64;
  half16 qa0 = load_a_frag(qrow, hi);        // K-dim 0..31
  half16 qa1 = load_a_frag(qrow + 32, hi);   // K-dim 32..63
  for (int nt = 0; nt < L / 16; ++nt) {
    const _Float16* krow =
        k16 + ((size_t)(bb * S_ + w0 + nt * 16 + l15)) * D_ + hh * 64;
    floatx8 acc = {};
    acc = wmma_f16(qa0, load_b_frag(krow, hi), acc);
    acc = wmma_f16(qa1, load_b_frag(krow + 32, hi), acc);
#pragma unroll
    for (int r = 0; r < 8; ++r)
      Sc[(wave * 16 + hi * 8 + r) * SSTR + nt * 16 + l15] =
          (_Float16)(acc[r] * 0.125f);
  }
  __syncthreads();

  // ---- phase 3: softmax numerator; fold 1/sum into PV epilogue -------------
  // 2 threads per row, half8-vectorized scans, shfl-combined.
  {
    const int row = threadIdx.x >> 1, half = threadIdx.x & 1;
    _Float16* rp = Sc + row * SSTR + half * (L / 2);
    const int seg = L / 2;  // multiple of 64 halves
    float mx = -1e30f;
    for (int j = 0; j < seg; j += 8) {
      half8 v8 = *(const half8*)(rp + j);
#pragma unroll
      for (int e = 0; e < 8; ++e) mx = fmaxf(mx, (float)v8[e]);
    }
    mx = fmaxf(mx, __shfl_xor(mx, 1, 32));  // combine the two halves
    float sum = 0.f;
    for (int j = 0; j < seg; j += 8) {
      half8 v8 = *(const half8*)(rp + j);
      half8 e8;
#pragma unroll
      for (int e = 0; e < 8; ++e) {
        float ev = __expf((float)v8[e] - mx);
        sum += ev;
        e8[e] = (_Float16)ev;
      }
      *(half8*)(rp + j) = e8;
    }
    sum += __shfl_xor(sum, 1, 32);
    if (half == 0) invs[row] = 1.f / sum;
  }
  __syncthreads();

  // ---- phase 4: O = (P @ V) * invsum  (A from probs LDS, B from Vt LDS) ----
  floatx8 acc[4] = {};
  const _Float16* prow = Sc + (wave * 16 + l15) * SSTR;
  for (int kk = 0; kk < L; kk += 32) {
    half16 a = load_a_frag(prow + kk, hi);
#pragma unroll
    for (int nt = 0; nt < 4; ++nt) {
      half16 bf = load_b_frag(Vt + (nt * 16 + l15) * VSTR + kk, hi);
      acc[nt] = wmma_f16(a, bf, acc[nt]);
    }
  }
#pragma unroll
  for (int r = 0; r < 8; ++r) {
    const int rowb = wave * 16 + hi * 8 + r;
    const float sc = invs[rowb];
    float* orow = attnpre + ((size_t)(bb * S_ + q0 + rowb)) * D_ + hh * 64;
#pragma unroll
    for (int nt = 0; nt < 4; ++nt) orow[nt * 16 + l15] = acc[nt][r] * sc;
  }
}

// ---------------- host side ----------------
static inline size_t alignup(size_t x) { return (x + 255) & ~(size_t)255; }

extern "C" void kernel_launch(void* const* d_in, const int* in_sizes, int n_in,
                              void* d_out, int out_size, void* d_ws,
                              size_t ws_size, hipStream_t stream) {
  (void)in_sizes; (void)n_in; (void)out_size; (void)ws_size;
  const float* x      = (const float*)d_in[0];
  // d_in[1] = mask (unused: window derived analytically)
  const float* ln_in_g = (const float*)d_in[2];
  const float* ln_in_b = (const float*)d_in[3];
  const float* wq = (const float*)d_in[4];
  const float* bq = (const float*)d_in[5];
  const float* wk = (const float*)d_in[6];
  const float* bk = (const float*)d_in[7];
  const float* wv = (const float*)d_in[8];
  const float* bv = (const float*)d_in[9];
  const float* ln1_g = (const float*)d_in[10];
  const float* ln1_b = (const float*)d_in[11];
  const float* w1 = (const float*)d_in[12];
  const float* b1 = (const float*)d_in[13];
  const float* w2 = (const float*)d_in[14];
  const float* b2 = (const float*)d_in[15];
  const float* ln2_g = (const float*)d_in[16];
  const float* ln2_b = (const float*)d_in[17];
  float* out = (float*)d_out;

  // workspace layout
  char* ws = (char*)d_ws;
  size_t off = 0;
  auto take = [&](size_t bytes) { char* p = ws + off; off = alignup(off + bytes); return p; };
  float*    h32     = (float*)   take((size_t)M_ * D_ * 4);
  _Float16* h16     = (_Float16*)take((size_t)M_ * D_ * 2);
  _Float16* q16     = (_Float16*)take((size_t)M_ * D_ * 2);
  _Float16* k16     = (_Float16*)take((size_t)M_ * D_ * 2);
  _Float16* v16     = (_Float16*)take((size_t)M_ * D_ * 2);
  float*    attnpre = (float*)   take((size_t)M_ * D_ * 4);
  float*    attnres = (float*)   take((size_t)M_ * D_ * 4);
  _Float16* o16     = (_Float16*)take((size_t)M_ * D_ * 2);
  _Float16* ffn1    = (_Float16*)take((size_t)M_ * FFN_ * 2);
  float*    ffn2    = (float*)   take((size_t)M_ * D_ * 4);
  _Float16* wqt     = (_Float16*)take((size_t)D_ * D_ * 2);
  _Float16* wkt     = (_Float16*)take((size_t)D_ * D_ * 2);
  _Float16* wvt     = (_Float16*)take((size_t)D_ * D_ * 2);
  _Float16* w1t     = (_Float16*)take((size_t)D_ * FFN_ * 2);
  _Float16* w2t     = (_Float16*)take((size_t)FFN_ * D_ * 2);

  // 1) weights -> f16, N-major
  {
    int nDD = D_ * D_, nDF = D_ * FFN_;
    k_convT<<<(nDD + 255) / 256, 256, 0, stream>>>(wq, wqt, D_, D_);
    k_convT<<<(nDD + 255) / 256, 256, 0, stream>>>(wk, wkt, D_, D_);
    k_convT<<<(nDD + 255) / 256, 256, 0, stream>>>(wv, wvt, D_, D_);
    k_convT<<<(nDF + 255) / 256, 256, 0, stream>>>(w1, w1t, D_, FFN_);
    k_convT<<<(nDF + 255) / 256, 256, 0, stream>>>(w2, w2t, FFN_, D_);
  }

  // 2) h = LN(x)  (f32 + f16 copies)
  k_add_ln<<<M_, 128, 0, stream>>>(x, nullptr, ln_in_g, ln_in_b, nullptr, h32,
                                   h16);

  // 3) q,k,v = h @ W + b   (f16 outputs, [B,S,D] with per-head column slices)
  dim3 gqkv(M_ / 32, D_ / 256);
  k_gemm_wmma<<<gqkv, 128, 0, stream>>>(h16, wqt, bq, nullptr, q16, M_, D_, D_);
  k_gemm_wmma<<<gqkv, 128, 0, stream>>>(h16, wkt, bk, nullptr, k16, M_, D_, D_);
  k_gemm_wmma<<<gqkv, 128, 0, stream>>>(h16, wvt, bv, nullptr, v16, M_, D_, D_);

  // 4) windowed attention: 512 blocks x 256 thr, ~293KB LDS (320KB/WG on CDNA5)
  (void)hipFuncSetAttribute((const void*)k_attn,
                            hipFuncAttributeMaxDynamicSharedMemorySize,
                            (int)ATTN_LDS);
  k_attn<<<B_ * H_ * (S_ / CHUNK_), 256, ATTN_LDS, stream>>>(q16, k16, v16,
                                                             attnpre);

  // 5) attn_out = attnpre + h ; o = LN1(attn_out) -> f16
  k_add_ln<<<M_, 128, 0, stream>>>(attnpre, h32, ln1_g, ln1_b, attnres, nullptr,
                                   o16);

  // 6) FFN: (o @ w1 + b1) @ w2 + b2
  dim3 g1(M_ / 32, FFN_ / 256);
  k_gemm_wmma<<<g1, 128, 0, stream>>>(o16, w1t, b1, nullptr, ffn1, M_, FFN_, D_);
  dim3 g2(M_ / 32, D_ / 256);
  k_gemm_wmma<<<g2, 128, 0, stream>>>(ffn1, w2t, b2, ffn2, nullptr, M_, D_,
                                      FFN_);

  // 7) out = LN2(ffn2 + attn_out)
  k_add_ln<<<M_, 128, 0, stream>>>(ffn2, attnres, ln2_g, ln2_b, nullptr, out,
                                   nullptr);
}
